// GRUMinimal_17471926960361
// MI455X (gfx1250) — compile-verified, matrix operand order checked by reading, not run on hardware
//
#include <hip/hip_runtime.h>
#include <math.h>

// ---------------------------------------------------------------------------
// GRU-minimal on MI455X (gfx1250): bf16 WMMA (v_wmma_f32_16x16x32_bf16),
// f32 accumulation, pre-packed B operands, async global->LDS B staging in the
// big input-projection GEMM, and a PERSISTENT kernel for the 512-step scan
// (device-wide barrier instead of 1024 kernel launches; h/F in registers).
// ---------------------------------------------------------------------------

typedef __attribute__((ext_vector_type(16))) __bf16 v16bf;
typedef __attribute__((ext_vector_type(8)))  __bf16 v8bf;
typedef __attribute__((ext_vector_type(8)))  float  v8f;
typedef __attribute__((ext_vector_type(4)))  float  v4f;

#define B_    64
#define T_    512
#define E_    1024
#define KT_   (E_ / 32)   // 32 k-tiles of 32
#define NT_   (E_ / 16)   // 64 n-tiles of 16
#define NB_   4           // n-tiles per wave (A-fragment reuse)
#define NBLK_ 8           // persistent-scan workgroups (64 waves total)

// 16-bit A-matrix element->K mapping (ISA 7.12.2):
// lane-half 0: e 0..7 -> K 0..7,  e 8..15 -> K 16..23
// lane-half 1: e 0..7 -> K 8..15, e 8..15 -> K 24..31
__device__ __forceinline__ int kmap(int e, int hl) {
    return e + 8 * hl + ((e >= 8) ? 8 : 0);
}

// ---------------------------------------------------------------------------
// Weight pre-pack: W[k][n] f32 row-major  ->  Wp[((kt*NT+nt)*32+lane)*16+e]
// bf16, so GEMM fragments are contiguous 32B per lane and whole (kt, 4-ntile)
// panels are contiguous 4KB blocks (async-to-LDS friendly).
// ---------------------------------------------------------------------------
__global__ void __launch_bounds__(256) pack_w(const float* __restrict__ W,
                                              __bf16* __restrict__ Wp) {
    int idx  = blockIdx.x * 256 + threadIdx.x;   // [0, E*E)
    int e    = idx & 15;
    int lane = (idx >> 4) & 31;
    int rest = idx >> 9;
    int nt   = rest & (NT_ - 1);
    int kt   = rest >> 6;
    int k    = kt * 32 + kmap(e, lane >> 4);
    int n    = nt * 16 + (lane & 15);
    Wp[idx]  = (__bf16)W[(size_t)k * E_ + n];
}

__global__ void zero_u32(unsigned* p) { *p = 0u; }

// ---------------------------------------------------------------------------
// Wave-level K-loop: bf16 row-major A x packed-B, NB_ accumulators, manual
// double-buffer so iteration k's WMMAs consume operands loaded at k-1.
// ---------------------------------------------------------------------------
__device__ __forceinline__ void wave_gemm_bf16A(const __bf16* __restrict__ A,
                                                int rowBase,
                                                const __bf16* __restrict__ Wp,
                                                int ntBase, int lane,
                                                v8f acc[NB_]) {
    const int hl = lane >> 4, ln = lane & 15;
    const __bf16* arow = A + (size_t)(rowBase + ln) * E_;
    v8bf c0 = *(const v8bf*)(arow + 8 * hl);
    v8bf c1 = *(const v8bf*)(arow + 16 + 8 * hl);
    v16bf b[NB_];
#pragma unroll
    for (int j = 0; j < NB_; ++j)
        b[j] = *(const v16bf*)(Wp + ((size_t)((ntBase + j) * 32 + lane) << 4));
    for (int kt = 0; kt < KT_; ++kt) {
        v16bf a;
#pragma unroll
        for (int i = 0; i < 8; ++i) { a[i] = c0[i]; a[8 + i] = c1[i]; }
        v16bf bb[NB_];
#pragma unroll
        for (int j = 0; j < NB_; ++j) bb[j] = b[j];
        if (kt + 1 < KT_) {   // issue next-iteration loads before the WMMAs
            c0 = *(const v8bf*)(arow + (kt + 1) * 32 + 8 * hl);
            c1 = *(const v8bf*)(arow + (kt + 1) * 32 + 16 + 8 * hl);
#pragma unroll
            for (int j = 0; j < NB_; ++j)
                b[j] = *(const v16bf*)(Wp +
                    ((size_t)(((kt + 1) * NT_ + ntBase + j) * 32 + lane) << 4));
        }
#pragma unroll
        for (int j = 0; j < NB_; ++j)
            acc[j] = __builtin_amdgcn_wmma_f32_16x16x32_bf16(
                false, a, false, bb[j], (short)0, acc[j], false, false);
    }
}

// ---------------------------------------------------------------------------
// Phase 1: XF = x@wxf + bf (f32, staged in d_out), XH = x@wxh + bh (bf16, ws).
// All 8 waves of a block share ntBase -> identical B panels, so stage each
// 4KB (kt, 4-ntile) panel into LDS via async global->LDS (ASYNCcnt-tracked),
// double-buffered; A (x rows, f32) is loaded/converted per-wave, pipelined.
// ---------------------------------------------------------------------------
__global__ void __launch_bounds__(256) gemm_x(const float* __restrict__ x,
                                              const __bf16* __restrict__ Wp,
                                              const float* __restrict__ bias,
                                              float* __restrict__ outF,
                                              __bf16* __restrict__ outH) {
    __shared__ __align__(32) __bf16 sB[2][NB_ * 512];   // 2 x 4KB panels
    const int tid  = threadIdx.x;
    const int lane = tid & 31, w = tid >> 5;
    const int mt = blockIdx.y * 8 + w;       // [0, 2048)
    const int ntBase = blockIdx.x * NB_;     // [0, 64) step 4
    const int hl = lane >> 4, ln = lane & 15;
    const float* arow = x + (size_t)(mt * 16 + ln) * E_;
    const unsigned long long wpbase = (unsigned long long)(size_t)Wp;

    {   // stage kt=0 panel into sB[0]: each of 256 threads moves 16B
        unsigned lds = (unsigned)(size_t)&sB[0][tid * 8];
        unsigned go  = (unsigned)(((ntBase * 512) + tid * 8) * 2);
        asm volatile("global_load_async_to_lds_b128 %0, %1, %2"
                     :: "v"(lds), "v"(go), "s"(wpbase) : "memory");
    }
    v4f f0 = *(const v4f*)(arow + 8 * hl);
    v4f f1 = *(const v4f*)(arow + 8 * hl + 4);
    v4f f2 = *(const v4f*)(arow + 16 + 8 * hl);
    v4f f3 = *(const v4f*)(arow + 16 + 8 * hl + 4);
    asm volatile("s_wait_asynccnt 0" ::: "memory");
    __syncthreads();

    v8f acc[NB_] = {};
    for (int kt = 0; kt < KT_; ++kt) {
        const int cur = kt & 1;
        if (kt + 1 < KT_) {   // async-stage next panel into the other buffer
            unsigned lds = (unsigned)(size_t)&sB[cur ^ 1][tid * 8];
            unsigned go  = (unsigned)(((((kt + 1) * NT_ + ntBase) * 512)
                                       + tid * 8) * 2);
            asm volatile("global_load_async_to_lds_b128 %0, %1, %2"
                         :: "v"(lds), "v"(go), "s"(wpbase) : "memory");
        }
        v16bf a;
#pragma unroll
        for (int i = 0; i < 4; ++i) {
            a[i]      = (__bf16)f0[i];
            a[4 + i]  = (__bf16)f1[i];
            a[8 + i]  = (__bf16)f2[i];
            a[12 + i] = (__bf16)f3[i];
        }
        if (kt + 1 < KT_) {   // pipeline next A chunk
            f0 = *(const v4f*)(arow + (kt + 1) * 32 + 8 * hl);
            f1 = *(const v4f*)(arow + (kt + 1) * 32 + 8 * hl + 4);
            f2 = *(const v4f*)(arow + (kt + 1) * 32 + 16 + 8 * hl);
            f3 = *(const v4f*)(arow + (kt + 1) * 32 + 16 + 8 * hl + 4);
        }
#pragma unroll
        for (int j = 0; j < NB_; ++j) {
            v16bf b = *(const v16bf*)&sB[cur][(j * 32 + lane) * 16];
            acc[j] = __builtin_amdgcn_wmma_f32_16x16x32_bf16(
                false, a, false, b, (short)0, acc[j], false, false);
        }
        asm volatile("s_wait_asynccnt 0" ::: "memory");
        __syncthreads();
    }

    if (outF) {
#pragma unroll
        for (int j = 0; j < NB_; ++j) {
            int n = (ntBase + j) * 16 + ln;
            float bv = bias[n];
#pragma unroll
            for (int r = 0; r < 8; ++r) {
                int m = mt * 16 + r + 8 * hl;
                outF[(size_t)m * E_ + n] = acc[j][r] + bv;
            }
        }
    } else {
#pragma unroll
        for (int j = 0; j < NB_; ++j) {
            int n = (ntBase + j) * 16 + ln;
            float bv = bias[n];
#pragma unroll
            for (int r = 0; r < 8; ++r) {
                int m = mt * 16 + r + 8 * hl;
                outH[(size_t)m * E_ + n] = (__bf16)(acc[j][r] + bv);
            }
        }
    }
}

// ---------------------------------------------------------------------------
// Device-wide sense barrier for the persistent scan (8 resident workgroups).
// Release own stores -> arrive (agent-scope atomic) -> spin with s_sleep ->
// acquire. The s_cluster_barrier is the CDNA5 fast path when dispatched as a
// workgroup cluster; ISA defines it as S_NOP when ClusterID==0 (our case).
// ---------------------------------------------------------------------------
__device__ __forceinline__ void grid_barrier(unsigned* cnt, unsigned target) {
    __threadfence();                 // make this wave's stores agent-visible
    __syncthreads();
    __builtin_amdgcn_s_cluster_barrier();   // NOP outside a cluster dispatch
    if (threadIdx.x == 0) {
        __hip_atomic_fetch_add(cnt, 1u, __ATOMIC_ACQ_REL,
                               __HIP_MEMORY_SCOPE_AGENT);
        while (__hip_atomic_load(cnt, __ATOMIC_ACQUIRE,
                                 __HIP_MEMORY_SCOPE_AGENT) < target) {
            __builtin_amdgcn_s_sleep(2);
        }
    }
    __syncthreads();
    __threadfence();                 // acquire: invalidate stale cached lines
}

// ---------------------------------------------------------------------------
// Persistent scan: one launch for all 512 steps. 8 blocks x 8 waves; each
// wave owns a fixed 16x64 strip (mt, 4 n-tiles) for the whole scan, so the
// f32 state h and gate F stay in registers; only bf16 operand mirrors
// (hbf, G) round-trip through L2 for the cross-wave WMMA reads.
//   gate:   F = sigmoid(XF_t + h@whf);  G = bf16(h*F)     [barrier]
//   update: p = tanh(XH_t + G@whh); h = (1-F)h + F p       [barrier]
// XF is staged f32 in outs[b][t][:] and overwritten by outs the same step.
// ---------------------------------------------------------------------------
__global__ void __launch_bounds__(256) scan_all(const float* __restrict__ h0,
                                                float* outs,  // XF in, outs out
                                                const __bf16* __restrict__ XH,
                                                __bf16* __restrict__ hbf,
                                                __bf16* __restrict__ G,
                                                const __bf16* __restrict__ whf_p,
                                                const __bf16* __restrict__ whh_p,
                                                float* __restrict__ hT,
                                                unsigned* cnt) {
    const int lane = threadIdx.x & 31, w = threadIdx.x >> 5;
    const int tile = blockIdx.x * 8 + w;          // [0, 64)
    const int mt = tile >> 4, ntBase = (tile & 15) * NB_;
    const int hl = lane >> 4, ln = lane & 15;

    // Load owned h0 elements into registers; publish bf16 mirror.
    float hv[NB_][8];
#pragma unroll
    for (int j = 0; j < NB_; ++j) {
        int n = (ntBase + j) * 16 + ln;
#pragma unroll
        for (int r = 0; r < 8; ++r) {
            int b = mt * 16 + r + 8 * hl;
            size_t o = (size_t)b * E_ + n;
            float v = h0[o];
            hv[j][r] = v;
            hbf[o] = (__bf16)v;
        }
    }
    unsigned bar = 0;
    grid_barrier(cnt, bar += NBLK_);

    for (int t = 0; t < T_; ++t) {
        // ---- gate phase: F = sigmoid(XF_t + h@whf), G = bf16(h*F) ----
        v8f acc[NB_] = {};
        wave_gemm_bf16A(hbf, mt * 16, whf_p, ntBase, lane, acc);
        float fv[NB_][8];
#pragma unroll
        for (int j = 0; j < NB_; ++j) {
            int n = (ntBase + j) * 16 + ln;
#pragma unroll
            for (int r = 0; r < 8; ++r) {
                int b = mt * 16 + r + 8 * hl;
                float pre = acc[j][r] + outs[((size_t)b * T_ + t) * E_ + n];
                float f = 1.f / (1.f + __expf(-pre));
                fv[j][r] = f;
                G[(size_t)b * E_ + n] = (__bf16)(hv[j][r] * f);
            }
        }
        grid_barrier(cnt, bar += NBLK_);

        // ---- update phase: p = tanh(XH_t + G@whh), h' = (1-F)h + F p ----
        v8f acc2[NB_] = {};
        wave_gemm_bf16A(G, mt * 16, whh_p, ntBase, lane, acc2);
#pragma unroll
        for (int j = 0; j < NB_; ++j) {
            int n = (ntBase + j) * 16 + ln;
#pragma unroll
            for (int r = 0; r < 8; ++r) {
                int b = mt * 16 + r + 8 * hl;
                size_t to = ((size_t)b * T_ + t) * E_ + n;
                float p = tanhf(acc2[j][r] + (float)XH[to]);
                float f = fv[j][r];
                float hn = (1.f - f) * hv[j][r] + f * p;
                hv[j][r] = hn;
                hbf[(size_t)b * E_ + n] = (__bf16)hn;
                outs[to] = hn;
            }
        }
        grid_barrier(cnt, bar += NBLK_);
    }

    // Final state hT straight from registers.
#pragma unroll
    for (int j = 0; j < NB_; ++j) {
        int n = (ntBase + j) * 16 + ln;
#pragma unroll
        for (int r = 0; r < 8; ++r) {
            int b = mt * 16 + r + 8 * hl;
            hT[(size_t)b * E_ + n] = hv[j][r];
        }
    }
}

// ---------------------------------------------------------------------------
// Host-side launch sequence (graph-capturable; everything on `stream`).
// Workspace layout (bytes):
//   [0,2M)    wxf packed bf16      [2M,4M)  wxh packed
//   [4M,6M)   whf packed           [6M,8M)  whh packed
//   [8M,72M)  XH bf16 [B][T][E]
//   [72M..)   hbf bf16 (128K) | G bf16 (128K) | barrier counter
// XF is staged f32 inside d_out's outs region and overwritten step-by-step.
// ---------------------------------------------------------------------------
extern "C" void kernel_launch(void* const* d_in, const int* in_sizes, int n_in,
                              void* d_out, int out_size, void* d_ws, size_t ws_size,
                              hipStream_t stream) {
    (void)in_sizes; (void)n_in; (void)out_size; (void)ws_size;
    const float* x   = (const float*)d_in[0];
    const float* h0  = (const float*)d_in[1];
    const float* wxf = (const float*)d_in[2];
    const float* wxh = (const float*)d_in[3];
    const float* bf  = (const float*)d_in[4];
    const float* whf = (const float*)d_in[5];
    const float* whh = (const float*)d_in[6];
    const float* bh  = (const float*)d_in[7];

    float* outs = (float*)d_out;                       // [B][T][E]
    float* hT   = outs + (size_t)B_ * T_ * E_;         // [B][E]

    char* ws = (char*)d_ws;
    __bf16* wxf_p = (__bf16*)(ws);
    __bf16* wxh_p = (__bf16*)(ws + (size_t)(2u << 20));
    __bf16* whf_p = (__bf16*)(ws + (size_t)(4u << 20));
    __bf16* whh_p = (__bf16*)(ws + (size_t)(6u << 20));
    __bf16* XH    = (__bf16*)(ws + (size_t)(8u << 20));
    char* tail = ws + (size_t)(72u << 20);
    __bf16*   hbf = (__bf16*)(tail);
    __bf16*   G   = (__bf16*)(tail + (128u << 10));
    unsigned* cnt = (unsigned*)(tail + (256u << 10));

    // Phase 0: pack weights to bf16 WMMA fragments; reset scan barrier.
    pack_w<<<(E_ * E_) / 256, 256, 0, stream>>>(wxf, wxf_p);
    pack_w<<<(E_ * E_) / 256, 256, 0, stream>>>(wxh, wxh_p);
    pack_w<<<(E_ * E_) / 256, 256, 0, stream>>>(whf, whf_p);
    pack_w<<<(E_ * E_) / 256, 256, 0, stream>>>(whh, whh_p);
    zero_u32<<<1, 1, 0, stream>>>(cnt);

    // Phase 1: big input projections (parallel over all timesteps).
    dim3 g1(NT_ / NB_, (B_ * T_ / 16) / 8);
    gemm_x<<<g1, 256, 0, stream>>>(x, wxf_p, bf, outs, nullptr);   // XF (f32)
    gemm_x<<<g1, 256, 0, stream>>>(x, wxh_p, bh, nullptr, XH);     // XH (bf16)

    // Phase 2: persistent scan — ONE launch for all 512 steps; weights
    // (4MB bf16 total) stay L2-resident, h/F stay in registers.
    scan_all<<<NBLK_, 256, 0, stream>>>(h0, outs, XH, hbf, G,
                                        whf_p, whh_p, hT, cnt);
}
